// QLSTMGen024_65481071405988
// MI455X (gfx1250) — compile-verified, matrix operand order checked by reading, not run on hardware
//
#include <hip/hip_runtime.h>
#include <cstdint>
#include <cstddef>

#define SEQ    512
#define BATCH  64
#define INDIM  512
#define HID    512
#define NG     2048                      // 4*HID
#define ROWS   (SEQ*BATCH)               // 32768

typedef __attribute__((ext_vector_type(16))) __bf16 v16bf;
typedef __attribute__((ext_vector_type(8)))  float  v8f;

union Frag { v16bf v; uint4 q[2]; };

// ---- workspace layout (bytes) ----
#define ZX_OFF     0ull
#define ZX_BYTES   ((unsigned long long)ROWS * NG * 4ull)        // 268 MB
#define WXB_OFF    (ZX_OFF + ZX_BYTES)
#define WB_BYTES   ((unsigned long long)NG * 512ull * 2ull)      // 2 MB each
#define WHB_OFF    (WXB_OFF + WB_BYTES)
#define HBUF_OFF   (WHB_OFF + WB_BYTES)
#define HBUF_BYTES (2ull * 64ull * 512ull * 2ull)                // double-buffered h (bf16)
#define BAR_OFF    (HBUF_OFF + HBUF_BYTES)

__device__ __forceinline__ unsigned short f2bf(float x) {
    unsigned u = __float_as_uint(x);
    u += 0x7FFFu + ((u >> 16) & 1u);     // round-to-nearest-even
    return (unsigned short)(u >> 16);
}
__device__ __forceinline__ float sigf(float x) {
    return 1.0f / (1.0f + __expf(-x));
}
__device__ __forceinline__ float tanh_fast(float x) {
    x = fminf(fmaxf(x, -12.0f), 12.0f);
    float e = __expf(2.0f * x);
    return (e - 1.0f) / (e + 1.0f);
}

// ---------------- weight conversion: fp32 (H, IN+H) x4 -> bf16 Wx(2048,512), Wh(2048,512)
__global__ void convert_weights(const float* __restrict__ Wf, const float* __restrict__ Wi,
                                const float* __restrict__ Wg, const float* __restrict__ Wo,
                                unsigned short* __restrict__ wxb, unsigned short* __restrict__ whb) {
    const int total = 4 * 512 * 1024;
    for (int idx = blockIdx.x * blockDim.x + threadIdx.x; idx < total;
         idx += gridDim.x * blockDim.x) {
        int g = idx >> 19;               // gate: 0..3
        int r = (idx >> 10) & 511;       // output neuron
        int c = idx & 1023;              // input col
        const float* W = (g == 0) ? Wf : (g == 1) ? Wi : (g == 2) ? Wg : Wo;
        unsigned short h = f2bf(W[(size_t)r * 1024 + c]);
        if (c < 512) wxb[((size_t)(g * 512 + r)) * 512 + c]         = h;
        else         whb[((size_t)(g * 512 + r)) * 512 + (c - 512)] = h;
    }
}

// ---------------- per-launch state reset (h0 = 0, barrier = 0); graph-replay safe
__global__ void init_state(unsigned* __restrict__ hbuf_u32, unsigned* __restrict__ bar) {
    const int n = (int)(HBUF_BYTES / 4);
    for (int i = blockIdx.x * blockDim.x + threadIdx.x; i < n; i += gridDim.x * blockDim.x)
        hbuf_u32[i] = 0u;
    if (blockIdx.x == 0 && threadIdx.x == 0) *bar = 0u;
}

// ---------------- Phase A: Zx[row][4H] = X @ Wx^T + b  (bf16 WMMA, fp32 acc)
// grid = (2048 row-tiles, 32 col-groups), block = 128 (4 waves; wave w -> col tile 4*gy+w)
__global__ void __launch_bounds__(128)
xproj_kernel(const float* __restrict__ X, const unsigned short* __restrict__ wxb,
             const float* __restrict__ bfv, const float* __restrict__ biv,
             const float* __restrict__ bgv, const float* __restrict__ bov,
             float* __restrict__ Zx) {
    __shared__ unsigned ldsA[16 * 256];          // 16 rows x 512 bf16 = 16 KB

    const int tid   = threadIdx.x;
    const int rtile = blockIdx.x;                // 0..2047 (16 rows of X each)

    // cooperative load + fp32->bf16 convert into LDS
    {
        const int row     = tid >> 3;            // 0..15
        const int colbase = (tid & 7) * 64;      // 0..448
        const float* src  = X + ((size_t)(rtile * 16 + row)) * 512 + colbase;
        unsigned* dst     = &ldsA[row * 256 + (colbase >> 1)];
#pragma unroll
        for (int j = 0; j < 32; ++j) {
            float2 f2 = ((const float2*)src)[j];
            dst[j] = (unsigned)f2bf(f2.x) | ((unsigned)f2bf(f2.y) << 16);
        }
    }
    __syncthreads();

    const int wave = tid >> 5, lane = tid & 31;
    const int nl = lane & 15, hiw = lane >> 4;
    const int ct = blockIdx.y * 4 + wave;        // 0..127
    const int n  = ct * 16 + nl;                 // 0..2047

    v8f acc = {};
#pragma unroll
    for (int kk = 0; kk < 512; kk += 32) {
        const int kb = kk + hiw * 8;
        Frag a, b;
        const uint4* ap = (const uint4*)&ldsA[nl * 256 + (kb >> 1)];
        a.q[0] = ap[0];  a.q[1] = ap[2];
        const uint4* bp = (const uint4*)(wxb + (size_t)n * 512 + kb);
        b.q[0] = bp[0];  b.q[1] = bp[2];
        acc = __builtin_amdgcn_wmma_f32_16x16x32_bf16(false, a.v, false, b.v,
                                                      (short)0, acc, false, false);
    }

    const float* bp = (n < 512) ? bfv : (n < 1024) ? biv : (n < 1536) ? bgv : bov;
    const float bias = bp[n & 511];
#pragma unroll
    for (int r = 0; r < 8; ++r) {
        const int grow = rtile * 16 + hiw * 8 + r;
        Zx[(size_t)grow * 2048 + n] = acc[r] + bias;
    }
}

// ---------------- Phase B: persistent recurrence. 32 blocks x 128 threads.
// block = (rt = batch row tile 0..3, hg = hid group 0..7); wave owns one 16x16 (b,hid)
// tile for ALL 4 gates; c-state lives in registers across all 512 steps.
__global__ void __launch_bounds__(128)
lstm_recurrent(const float* __restrict__ Zx, const unsigned short* __restrict__ whb,
               unsigned short* __restrict__ hbuf, unsigned* __restrict__ bar,
               float* __restrict__ out) {
    __shared__ unsigned ldsA[16 * 256];          // 16 x 512 bf16 h-tile

    const int tid  = threadIdx.x;
    const int rt   = blockIdx.x & 3;             // batch tile
    const int hg   = blockIdx.x >> 2;            // hid group
    const int wave = tid >> 5, lane = tid & 31;
    const int nl = lane & 15, hiw = lane >> 4;
    const int n  = hg * 64 + wave * 16 + nl;     // hid col 0..511

    const int ldrow = tid >> 3;
    const int ldcol = (tid & 7) * 64;

    const unsigned short* wb0 = whb + ((size_t)n)          * 512;
    const unsigned short* wb1 = whb + ((size_t)n + 512)    * 512;
    const unsigned short* wb2 = whb + ((size_t)n + 1024)   * 512;
    const unsigned short* wb3 = whb + ((size_t)n + 1536)   * 512;

    float cst[8], hlast[8];
#pragma unroll
    for (int r = 0; r < 8; ++r) { cst[r] = 0.0f; hlast[r] = 0.0f; }

    unsigned target = 0;

    for (int t = 0; t < SEQ; ++t) {
        // stage h_{t} (bf16) rows rt*16..rt*16+15 into LDS
        {
            const uint4* src = (const uint4*)(hbuf + (size_t)(t & 1) * 32768 +
                                              (size_t)(rt * 16 + ldrow) * 512 + ldcol);
            uint4* dst = (uint4*)&ldsA[ldrow * 256 + (ldcol >> 1)];
#pragma unroll
            for (int j = 0; j < 8; ++j) dst[j] = src[j];
        }
        __syncthreads();

        // init accumulators from precomputed x-projection (bias included)
        v8f a0 = {}, a1 = {}, a2 = {}, a3 = {};
        {
            const size_t rbase = (size_t)t * 64 + rt * 16 + hiw * 8;
#pragma unroll
            for (int r = 0; r < 8; ++r) {
                const float* zr = Zx + (rbase + r) * 2048 + n;
                a0[r] = zr[0];  a1[r] = zr[512];  a2[r] = zr[1024];  a3[r] = zr[1536];
            }
        }

        // z += h_t @ Wh^T : 16 K-chunks x 4 gates = 64 WMMAs
#pragma unroll
        for (int kk = 0; kk < 512; kk += 32) {
            const int kb = kk + hiw * 8;
            Frag a;
            const uint4* ap = (const uint4*)&ldsA[nl * 256 + (kb >> 1)];
            a.q[0] = ap[0];  a.q[1] = ap[2];

            Frag b0, b1, b2, b3;
            const uint4* p0 = (const uint4*)(wb0 + kb);
            const uint4* p1 = (const uint4*)(wb1 + kb);
            const uint4* p2 = (const uint4*)(wb2 + kb);
            const uint4* p3 = (const uint4*)(wb3 + kb);
            b0.q[0] = p0[0]; b0.q[1] = p0[2];
            b1.q[0] = p1[0]; b1.q[1] = p1[2];
            b2.q[0] = p2[0]; b2.q[1] = p2[2];
            b3.q[0] = p3[0]; b3.q[1] = p3[2];

            a0 = __builtin_amdgcn_wmma_f32_16x16x32_bf16(false, a.v, false, b0.v, (short)0, a0, false, false);
            a1 = __builtin_amdgcn_wmma_f32_16x16x32_bf16(false, a.v, false, b1.v, (short)0, a1, false, false);
            a2 = __builtin_amdgcn_wmma_f32_16x16x32_bf16(false, a.v, false, b2.v, (short)0, a2, false, false);
            a3 = __builtin_amdgcn_wmma_f32_16x16x32_bf16(false, a.v, false, b3.v, (short)0, a3, false, false);
        }

        // gating + state update + outputs
        {
            const int bidx0 = rt * 16 + hiw * 8;
            float* outp = out + (size_t)t * 32768;
            unsigned short* hnext = hbuf + (size_t)((t + 1) & 1) * 32768;
#pragma unroll
            for (int r = 0; r < 8; ++r) {
                float f  = sigf(a0[r]);
                float ii = sigf(a1[r]);
                float g  = tanh_fast(a2[r]);
                float oo = sigf(a3[r]);
                float cn = f * cst[r] + ii * g;
                cst[r] = cn;
                float h = oo * tanh_fast(cn);
                hlast[r] = h;
                const size_t bi_ = (size_t)(bidx0 + r);
                outp[bi_ * 512 + n]  = h;
                hnext[bi_ * 512 + n] = f2bf(h);
            }
        }

        // grid-wide step barrier (counting; counter reset each launch by init_state)
        target += 32;
        __threadfence();                 // release h stores
        __syncthreads();
        if (tid == 0) {
            __hip_atomic_fetch_add(bar, 1u, __ATOMIC_RELAXED, __HIP_MEMORY_SCOPE_AGENT);
            while (__hip_atomic_load(bar, __ATOMIC_RELAXED, __HIP_MEMORY_SCOPE_AGENT) < target)
                __builtin_amdgcn_s_sleep(2);
        }
        __syncthreads();
        __threadfence();                 // acquire other blocks' h stores
    }

    // final (hx, cx)
    {
        const int bidx0 = rt * 16 + hiw * 8;
        float* hx = out + (size_t)SEQ * 32768;
        float* cx = hx + 32768;
#pragma unroll
        for (int r = 0; r < 8; ++r) {
            hx[(size_t)(bidx0 + r) * 512 + n] = hlast[r];
            cx[(size_t)(bidx0 + r) * 512 + n] = cst[r];
        }
    }
}

extern "C" void kernel_launch(void* const* d_in, const int* in_sizes, int n_in,
                              void* d_out, int out_size, void* d_ws, size_t ws_size,
                              hipStream_t stream) {
    const float* X   = (const float*)d_in[0];
    const float* Wf  = (const float*)d_in[1];
    const float* bf_ = (const float*)d_in[2];
    const float* Wi  = (const float*)d_in[3];
    const float* bi_ = (const float*)d_in[4];
    const float* Wg  = (const float*)d_in[5];
    const float* bg_ = (const float*)d_in[6];
    const float* Wo  = (const float*)d_in[7];
    const float* bo_ = (const float*)d_in[8];
    float* out = (float*)d_out;

    char* ws = (char*)d_ws;
    float*          Zx   = (float*)(ws + ZX_OFF);
    unsigned short* wxb  = (unsigned short*)(ws + WXB_OFF);
    unsigned short* whb  = (unsigned short*)(ws + WHB_OFF);
    unsigned short* hbuf = (unsigned short*)(ws + HBUF_OFF);
    unsigned*       bar  = (unsigned*)(ws + BAR_OFF);

    convert_weights<<<2048, 256, 0, stream>>>(Wf, Wi, Wg, Wo, wxb, whb);
    init_state<<<64, 256, 0, stream>>>((unsigned*)(ws + HBUF_OFF), bar);
    xproj_kernel<<<dim3(2048, 32), 128, 0, stream>>>(X, wxb, bf_, bi_, bg_, bo_, Zx);
    lstm_recurrent<<<32, 128, 0, stream>>>(Zx, whb, hbuf, bar, out);
}
